// MS_SSIM_L1_LOSS_56264071577945
// MI455X (gfx1250) — matrix operand order repaired and checked by reading, not compiled, not run on hardware
//
#include <hip/hip_runtime.h>
#include <hip/hip_bf16.h>
#include <math.h>

typedef __attribute__((ext_vector_type(16))) _Float16 v16h;
typedef __attribute__((ext_vector_type(8)))  _Float16 v8h;
typedef __attribute__((ext_vector_type(8)))  short    v8s;
typedef __attribute__((ext_vector_type(8)))  float    v8f;

#define NC     7
#define SIG_N  5
#define B_     16
#define H_     256
#define W_     256
#define PAD_   16
#define C1_    1.0e-4f
#define C2_    9.0e-4f
#define ALPHA_ 0.84f
#define COMP_  200.0f

#define XSTR 296   // padded LDS stride (halfs) for the 288-wide x/y staging rows
#define RCS  56    // padded LDS stride (halfs) for the 48-deep RC columns

static __device__ __forceinline__ v8f zero8() {
  v8f z;
#pragma unroll
  for (int i = 0; i < 8; ++i) z[i] = 0.0f;
  return z;
}

static __device__ __forceinline__ v8h habs8(v8h v) {
  v8s u = __builtin_bit_cast(v8s, v);
  u = u & (short)0x7FFF;                 // clear f16 sign bits
  return __builtin_bit_cast(v8h, u);
}

// Packed-f16 quantity: x, y, x*x, y*y, x*y, |x-y| (v_pk_* ops).
template <int Q>
static __device__ __forceinline__ v8h quant8(v8h x, v8h y) {
  if      (Q == 0) return x;
  else if (Q == 1) return y;
  else if (Q == 2) return x * x;
  else if (Q == 3) return y * y;
  else if (Q == 4) return x * y;
  else             return habs8(x - y);
}

static __device__ __forceinline__ v16h cat16(v8h lo, v8h hi) {
  v16h r;
#pragma unroll
  for (int e = 0; e < 8; ++e) { r[e] = lo[e]; r[e + 8] = hi[e]; }
  return r;
}

// Extract the separable 1-D gaussians: g2d = outer(g,g), sum(g)=1
// => g[j] = g2d[16][j] / sqrt(g2d[16][16]).
__global__ void g1d_extract_kernel(const float* __restrict__ g, float* __restrict__ g1d) {
  int t = threadIdx.x;
  if (t < SIG_N * 33) {
    int s = t / 33, k = t % 33;
    const float* plane = g + (size_t)(NC * s) * 33 * 33;   // class 0 of sigma s
    float center = plane[16 * 33 + 16];
    g1d[t] = plane[16 * 33 + k] / sqrtf(center);
  }
}

// Separable conv of quantity Q over one 16x256 strip:
// row pass (48 tiles across 8 waves) -> transposed f16 RC in LDS -> col pass
// (2 tiles per wave). Banded-Toeplitz A: out[m] = sum_j A[m,j] src[j],
// A[m,j] = g[j-m], j in [0,48), K padded to 64 -> 2x V_WMMA_F32_16X16X32_F16.
// A1 is zero for K>=16, so the upper-half-wave chunk-1 B rows (K=16..31) are
// don't-care: load them unconditionally (finite LDS data), no exec masking.
template <int Q>
static __device__ __forceinline__ void conv_tiles(
    const _Float16* __restrict__ xbuf, const _Float16* __restrict__ ybuf,
    _Float16* __restrict__ rc, v16h a0, v16h a1,
    int wave, int lm, int off8, int boff, v8f out[2])
{
  __syncthreads();                 // RC reuse fence (prev col pass done)
  // ---------- row pass: unit u = wave + 8*i ; wt = u%16, rg = u/16 ----------
#pragma unroll
  for (int i = 0; i < 6; ++i) {
    const int wt  = wave + 8 * (i & 1);
    const int rg  = i >> 1;
    const int row = rg * 16 + lm;            // N = source row within group
    const _Float16* xr = xbuf + row * XSTR + wt * 16;
    const _Float16* yr = ybuf + row * XSTR + wt * 16;
    v8h xa = *(const v8h*)(xr + boff);
    v8h xb = *(const v8h*)(xr + boff + 8);
    v8h xc = *(const v8h*)(xr + 32);
    v8h xd = *(const v8h*)(xr + 40);
    v8h ya = *(const v8h*)(yr + boff);
    v8h yb = *(const v8h*)(yr + boff + 8);
    v8h yc = *(const v8h*)(yr + 32);
    v8h yd = *(const v8h*)(yr + 40);
    v16h b0 = cat16(quant8<Q>(xa, ya), quant8<Q>(xb, yb));
    v16h b1 = cat16(quant8<Q>(xc, yc), quant8<Q>(xd, yd));
    v8f cc = zero8();
    cc = __builtin_amdgcn_wmma_f32_16x16x32_f16(false, a0, false, b0, (short)0, cc, false, false);
    cc = __builtin_amdgcn_wmma_f32_16x16x32_f16(false, a1, false, b1, (short)0, cc, false, false);
    // C elem (r,lane): M = w-in-tile = r+off8, N = row -> store transposed f16
#pragma unroll
    for (int r = 0; r < 8; ++r)
      rc[(wt * 16 + r + off8) * RCS + row] = (_Float16)cc[r];
  }
  __syncthreads();
  // ---------- column pass: 2 tiles per wave ----------
#pragma unroll
  for (int t = 0; t < 2; ++t) {
    const int wt = wave * 2 + t;
    const _Float16* cb = rc + (wt * 16 + lm) * RCS;   // N = w = wt*16+lm
    v8h r0 = *(const v8h*)(cb + boff);
    v8h r1 = *(const v8h*)(cb + boff + 8);
    v8h r2 = *(const v8h*)(cb + 32);
    v8h r3 = *(const v8h*)(cb + 40);
    v16h b0 = cat16(r0, r1);
    v16h b1 = cat16(r2, r3);
    v8f acc = zero8();
    acc = __builtin_amdgcn_wmma_f32_16x16x32_f16(false, a0, false, b0, (short)0, acc, false, false);
    acc = __builtin_amdgcn_wmma_f32_16x16x32_f16(false, a1, false, b1, (short)0, acc, false, false);
    out[t] = acc;
  }
}

// One workgroup = one 16x256 output strip of one (b,c) channel.
__global__ __launch_bounds__(256)
void MS_SSIM_L1_LOSS_56264071577945_kernel(
    const float* __restrict__ xg, const float* __restrict__ yg,
    const float* __restrict__ g1d,
    float* __restrict__ contrib, float* __restrict__ l1plane)
{
  extern __shared__ char smem_raw[];
  _Float16* xbuf = (_Float16*)smem_raw;        // [48][XSTR]
  _Float16* ybuf = xbuf + 48 * XSTR;           // [48][XSTR]
  _Float16* rc   = ybuf + 48 * XSTR;           // [256][RCS]  (w-major, transposed)

  const int ch    = blockIdx.x >> 4;           // b*NC + c
  const int strip = blockIdx.x & 15;
  const int hbase = strip * 16;
  const int tid   = threadIdx.x;
  const int lane  = tid & 31;
  const int wave  = tid >> 5;
  const int lm    = lane & 15;                 // M (A) / N (B,C) index
  const int off8  = (lane >> 4) * 8;           // C/D M-offset (A K-offset is 2*off8)
  const int boff  = (lane >> 4) * 16;          // B K-offset

  const float* xc = xg + (size_t)ch * (H_ * W_);
  const float* yc = yg + (size_t)ch * (H_ * W_);

  // ---- stage zero-padded inputs as f16 into LDS (division-free) ----
#pragma unroll
  for (int i = 0; i < 6; ++i) {
    const int r    = wave * 6 + i;             // 8 waves * 6 = 48 rows
    const int grow = hbase - PAD_ + r;
    const bool rok = (grow >= 0) && (grow < H_);
    const float* xrow = xc + grow * W_;
    const float* yrow = yc + grow * W_;
#pragma unroll
    for (int j = 0; j < 9; ++j) {
      const int cidx = lane + 32 * j;          // 0..287
      const int gcol = cidx - PAD_;
      const bool ok  = rok && (gcol >= 0) && (gcol < W_);
      float xv = ok ? xrow[gcol] : 0.0f;
      float yv = ok ? yrow[gcol] : 0.0f;
      xbuf[r * XSTR + cidx] = (_Float16)xv;
      ybuf[r * XSTR + cidx] = (_Float16)yv;
    }
  }

  v8f csprod[2];
#pragma unroll
  for (int t = 0; t < 2; ++t)
#pragma unroll
    for (int i = 0; i < 8; ++i) csprod[t][i] = 1.0f;

  for (int s = 0; s < SIG_N; ++s) {
    // ---- banded Toeplitz weight fragments for this sigma ----
    // A K-map (contiguous, matches B): K = e + 2*off8.
    v16h a0, a1;
#pragma unroll
    for (int e = 0; e < 16; ++e) {
      int k = e + 2 * off8;
      float v0 = (k >= lm) ? g1d[s * 33 + (k - lm)]      : 0.0f;  // taps j-m in [0,32]
      float v1 = (k <= lm) ? g1d[s * 33 + (32 + k - lm)] : 0.0f;  // window rows 32..47
      a0[e] = (_Float16)v0;
      a1[e] = (_Float16)v1;
    }

    v8f mux[2], muy[2], exx[2], eyy[2], exy[2], l1f[2];
    conv_tiles<0>(xbuf, ybuf, rc, a0, a1, wave, lm, off8, boff, mux);
    conv_tiles<1>(xbuf, ybuf, rc, a0, a1, wave, lm, off8, boff, muy);
    conv_tiles<2>(xbuf, ybuf, rc, a0, a1, wave, lm, off8, boff, exx);
    conv_tiles<3>(xbuf, ybuf, rc, a0, a1, wave, lm, off8, boff, eyy);
    conv_tiles<4>(xbuf, ybuf, rc, a0, a1, wave, lm, off8, boff, exy);
    if (s == SIG_N - 1)                         // uniform branch
      conv_tiles<5>(xbuf, ybuf, rc, a0, a1, wave, lm, off8, boff, l1f);

    // ---------- combine: cs product across sigmas; l, L1 on the last sigma ----------
#pragma unroll
    for (int t = 0; t < 2; ++t) {
#pragma unroll
      for (int i = 0; i < 8; ++i) {
        float mx  = mux[t][i], my = muy[t][i];
        float sxx = exx[t][i] - mx * mx;
        float syy = eyy[t][i] - my * my;
        float sxy = exy[t][i] - mx * my;
        float cs  = (2.0f * sxy + C2_) / (sxx + syy + C2_);
        csprod[t][i] *= cs;
        if (s == SIG_N - 1) {
          float l = (2.0f * mx * my + C1_) / (mx * mx + my * my + C1_);
          int h = hbase + i + off8;
          int w = (wave * 2 + t) * 16 + lm;
          size_t o = (size_t)ch * (H_ * W_) + (size_t)h * W_ + w;
          contrib[o] = l * csprod[t][i];
          l1plane[o] = l1f[t][i];
        }
      }
    }
  }
}

// Per-pixel cross-class product + L1 mean, block tree reduction (deterministic).
__global__ __launch_bounds__(256)
void combine_reduce_kernel(const float* __restrict__ contrib,
                           const float* __restrict__ l1plane,
                           float* __restrict__ partial)
{
  __shared__ float red[256];
  int idx = blockIdx.x * 256 + threadIdx.x;     // over B*H*W
  int b   = idx >> 16;
  int pix = idx & 65535;
  float prod = 1.0f, l1s = 0.0f;
#pragma unroll
  for (int c = 0; c < NC; ++c) {
    size_t o = ((size_t)(b * NC + c) << 16) + pix;
    prod *= contrib[o];
    l1s  += l1plane[o];
  }
  float val = ALPHA_ * (1.0f - prod) + (1.0f - ALPHA_) * (l1s * (1.0f / NC));
  red[threadIdx.x] = val;
  __syncthreads();
  for (int off = 128; off > 0; off >>= 1) {
    if (threadIdx.x < off) red[threadIdx.x] += red[threadIdx.x + off];
    __syncthreads();
  }
  if (threadIdx.x == 0) partial[blockIdx.x] = red[0];
}

__global__ __launch_bounds__(256)
void finish_kernel(const float* __restrict__ partial, int n, float* __restrict__ out)
{
  __shared__ float red[256];
  float a = 0.0f;
  for (int i = threadIdx.x; i < n; i += 256) a += partial[i];
  red[threadIdx.x] = a;
  __syncthreads();
  for (int off = 128; off > 0; off >>= 1) {
    if (threadIdx.x < off) red[threadIdx.x] += red[threadIdx.x + off];
    __syncthreads();
  }
  if (threadIdx.x == 0) out[0] = COMP_ * red[0] / (float)(B_ * H_ * W_);
}

extern "C" void kernel_launch(void* const* d_in, const int* in_sizes, int n_in,
                              void* d_out, int out_size, void* d_ws, size_t ws_size,
                              hipStream_t stream)
{
  (void)in_sizes; (void)n_in; (void)out_size; (void)ws_size;
  const float* x = (const float*)d_in[0];
  const float* y = (const float*)d_in[1];
  const float* g = (const float*)d_in[2];
  float* ws = (float*)d_ws;

  const size_t PLANE = (size_t)B_ * NC * H_ * W_;   // 7,340,032 floats
  float* g1d     = ws;                              // 256 floats reserved
  float* contrib = ws + 256;
  float* l1pl    = contrib + PLANE;
  float* partial = l1pl + PLANE;                    // 4096 floats
  float* out     = (float*)d_out;

  g1d_extract_kernel<<<1, 256, 0, stream>>>(g, g1d);

  dim3 gridA(B_ * NC * (H_ / 16));                  // 1792 strips
  size_t shmem = (size_t)(2 * 48 * XSTR + 256 * RCS) * 2;   // f16 bytes: 85,504
  MS_SSIM_L1_LOSS_56264071577945_kernel<<<gridA, 256, shmem, stream>>>(
      x, y, g1d, contrib, l1pl);

  int nblk = (B_ * H_ * W_) / 256;                  // 4096
  combine_reduce_kernel<<<nblk, 256, 0, stream>>>(contrib, l1pl, partial);
  finish_kernel<<<1, 256, 0, stream>>>(partial, nblk, out);
}
